// Ge2eLoss_9904194584676
// MI455X (gfx1250) — matrix operand order repaired and checked by reading, not compiled, not run on hardware
//
#include <hip/hip_runtime.h>
#include <hip/hip_bf16.h>

// GE2E loss: N=2048 speakers, M=16 utterances, D=256.
// loss = mean over rows ji of ( logsumexp_k(w*dot(e_ji, c_k)+b) - (w*diag_ji+b) )
// where column j (own speaker) is replaced by the excluded-self dot diag_ji.

typedef __attribute__((ext_vector_type(16))) __bf16 v16bf;
typedef __attribute__((ext_vector_type(8)))  float  v8f;
typedef __attribute__((ext_vector_type(4)))  unsigned int tdm_v4u;
typedef __attribute__((ext_vector_type(8)))  int          tdm_v8i;
typedef __attribute__((ext_vector_type(4)))  int          tdm_v4i;

#define N_SPK 2048
#define D_DIM 256
#define NM    32768
#define NCT   128   // column tiles of 16 centroids
#define NKT   8     // K tiles of 32

// workspace layout (bytes)
#define WS_SUMS 0                                  // 2048*256 f32   = 2 MB
#define WS_DIAG (N_SPK * D_DIM * 4)                // 32768 f32      = 128 KB
#define WS_PART (WS_DIAG + NM * 4)                 // 2048 f32       = 8 KB
#define WS_BPK  (WS_PART + N_SPK * 4)              // packed bf16 B  = 2 MB

#if __has_builtin(__builtin_amdgcn_tensor_load_to_lds)
#define GE2E_USE_TDM 1
#else
#define GE2E_USE_TDM 0
#endif

// ---------------------------------------------------------------- sums over M
__global__ __launch_bounds__(256) void k_sums(const float* __restrict__ emb,
                                              float* __restrict__ sums) {
  const int n = blockIdx.x;          // speaker
  const int d = threadIdx.x;         // feature
  const float* p = emb + (size_t)n * 16 * D_DIM + d;
  float s = 0.f;
#pragma unroll
  for (int i = 0; i < 16; ++i) s += p[i * D_DIM];
  sums[n * D_DIM + d] = s;
}

// ------------------------------------------- pack centroids into B fragments
// Fragment layout (bf16 element index):
//   idx = ((ct*8 + kt)*2 + term)*512 + L*16 + j
// lane L holds column n = ct*16 + (L&15), K = kt*32 + (j>=8?16:0) + (L>=16?8:0) + (j&7)
// term 0 = bf16(hi), term 1 = bf16(residual)
__global__ __launch_bounds__(256) void k_pack(const float* __restrict__ sums,
                                              __bf16* __restrict__ bpack) {
  const int idx  = blockIdx.x * 256 + threadIdx.x;   // 0 .. 2^20-1
  const int j    = idx & 15;
  const int L    = (idx >> 4) & 31;
  const int term = (idx >> 9) & 1;
  const int kt   = (idx >> 10) & 7;
  const int ct   = idx >> 13;
  const int n    = ct * 16 + (L & 15);
  const int k    = kt * 32 + ((j >> 3) << 4) + ((L >> 4) << 3) + (j & 7);
  const float c  = sums[n * D_DIM + k] * (1.0f / 16.0f);
  const __bf16 h = (__bf16)c;
  const __bf16 l = (__bf16)(c - (float)h);
  bpack[idx] = term ? l : h;
}

// -------------------------------------------------- leave-one-out diag dots
__global__ __launch_bounds__(256) void k_diag(const float* __restrict__ emb,
                                              const float* __restrict__ sums,
                                              float* __restrict__ diag) {
  const int gid  = blockIdx.x * 256 + threadIdx.x;
  const int row  = gid >> 5;                 // one wave per row, exactly 32768
  const int lane = threadIdx.x & 31;
  const int spk  = row >> 4;
  const float* e = emb + (size_t)row * D_DIM + lane * 8;
  const float* s = sums + spk * D_DIM + lane * 8;
  float4 ea = *(const float4*)(e);
  float4 eb = *(const float4*)(e + 4);
  float4 sa = *(const float4*)(s);
  float4 sb = *(const float4*)(s + 4);
  float acc = (sa.x - ea.x) * ea.x + (sa.y - ea.y) * ea.y +
              (sa.z - ea.z) * ea.z + (sa.w - ea.w) * ea.w +
              (sb.x - eb.x) * eb.x + (sb.y - eb.y) * eb.y +
              (sb.z - eb.z) * eb.z + (sb.w - eb.w) * eb.w;
#pragma unroll
  for (int off = 16; off; off >>= 1) acc += __shfl_xor(acc, off);
  if (lane == 0) diag[row] = acc * (1.0f / 15.0f);
}

#if GE2E_USE_TDM
// Issue one TDM descriptor copying a contiguous 16 KB column tile into LDS.
// D# per CDNA5 ISA ch.8: 1-D tile of 2048 x 8-byte elements.
__device__ __forceinline__ void tdm_load_tile(const __bf16* gsrc,
                                              unsigned ldsByteAddr) {
  const unsigned long long ga = (unsigned long long)(size_t)gsrc;
  tdm_v4u g0;
  g0[0] = 1u;                                         // count=1, user mode
  g0[1] = ldsByteAddr;                                // lds_addr
  g0[2] = (unsigned)(ga & 0xFFFFFFFFu);               // global_addr[31:0]
  g0[3] = (unsigned)((ga >> 32) & 0x01FFFFFFu)        // global_addr[56:32]
          | (2u << 30);                               // type=2 ("image")
  tdm_v8i g1;
  g1[0] = (int)(3u << 16);        // wg_mask=0, data_size=3 (8 B), no extras
  g1[1] = (int)(2048u << 16);     // tensor_dim0 lo16 -> [31:16]  (=2048)
  g1[2] = (int)(1u << 16);        // tensor_dim0 hi16=0 | tensor_dim1 lo16=1
  g1[3] = (int)(2048u << 16);     // tensor_dim1 hi16=0 | tile_dim0=2048
  g1[4] = 1;                      // tile_dim1=1, tile_dim2=0
  g1[5] = 2048;                   // tensor_dim0_stride lo32
  g1[6] = (int)(2048u << 16);     // stride0 hi16=0 | tensor_dim1_stride lo16
  g1[7] = 0;
  const tdm_v4i gz = {0, 0, 0, 0};
#if defined(__clang_major__) && (__clang_major__ >= 23)
  const tdm_v8i gz8 = {0, 0, 0, 0, 0, 0, 0, 0};
  __builtin_amdgcn_tensor_load_to_lds(g0, g1, gz, gz, gz8, 0);
#else
  __builtin_amdgcn_tensor_load_to_lds(g0, g1, gz, gz, 0);
#endif
}
#endif

// --------------------------------- main: bf16x3 WMMA GEMM + online logsumexp
__global__ __launch_bounds__(256) void k_main(const float* __restrict__ emb,
                                              const __bf16* __restrict__ bpack,
                                              const float* __restrict__ diag,
                                              const float* __restrict__ wp,
                                              const float* __restrict__ bp,
                                              float* __restrict__ part) {
  __shared__ uint4 ldsb[2][1024];            // 2 x 16 KB B-tile double buffer
  const int tid  = threadIdx.x;
  const int wave = tid >> 5;
  const int lane = tid & 31;
  const int half = lane >> 4;                // 0: lanes 0-15, 1: lanes 16-31
  const int ln   = lane & 15;
  const int rowTile = blockIdx.x * 8 + wave; // 0..2047 (== label column)
  const int rowBase = rowTile << 4;
  const float w = wp[0], b = bp[0];

#if GE2E_USE_TDM
  // kick off async DMA of tile 0 while we load/split A
  if (wave == 0) tdm_load_tile(bpack, (unsigned)(size_t)(void*)&ldsb[0][0]);
#endif

  // ---- load this wave's 16x256 A block, split into bf16 hi/lo fragments ----
  // A layout (16-bit, 16x32): lane L -> row m = L&15;
  //   VGPR 0..3: K = kt*32 + half*8 + {0..7};  VGPR 4..7: K = kt*32 + 16 + half*8 + {0..7}
  v16bf ah[NKT], al[NKT];
  {
    const float* erow = emb + (size_t)(rowBase + ln) * D_DIM;
#pragma unroll
    for (int kt = 0; kt < NKT; ++kt) {
      const int k0 = kt * 32 + half * 8;
      const int k1 = kt * 32 + 16 + half * 8;
      float xs[16];
      float4 t;
      t = *(const float4*)(erow + k0);     xs[0]=t.x; xs[1]=t.y; xs[2]=t.z; xs[3]=t.w;
      t = *(const float4*)(erow + k0 + 4); xs[4]=t.x; xs[5]=t.y; xs[6]=t.z; xs[7]=t.w;
      t = *(const float4*)(erow + k1);     xs[8]=t.x; xs[9]=t.y; xs[10]=t.z; xs[11]=t.w;
      t = *(const float4*)(erow + k1 + 4); xs[12]=t.x; xs[13]=t.y; xs[14]=t.z; xs[15]=t.w;
#pragma unroll
      for (int j = 0; j < 16; ++j) {
        const __bf16 h = (__bf16)xs[j];
        ah[kt][j] = h;
        al[kt][j] = (__bf16)(xs[j] - (float)h);
      }
    }
  }

  // diag for this lane's 8 rows (C layout: VGPR v -> row m = half*8 + v)
  float dg[8];
#pragma unroll
  for (int v = 0; v < 8; ++v) dg[v] = diag[rowBase + half * 8 + v];

  float mrun[8], srun[8];
#pragma unroll
  for (int v = 0; v < 8; ++v) { mrun[v] = -__builtin_inff(); srun[v] = 0.f; }

#if GE2E_USE_TDM
  if (wave == 0) __builtin_amdgcn_s_wait_tensorcnt(0);
#else
  {
    const uint4* s = (const uint4*)bpack + tid * 4;
    uint4 r0 = s[0], r1 = s[1], r2 = s[2], r3 = s[3];
    uint4* d = &ldsb[0][tid * 4];
    d[0] = r0; d[1] = r1; d[2] = r2; d[3] = r3;
  }
#endif
  __syncthreads();

  const int labCT = rowTile >> 4;
  const int labLn = rowTile & 15;

#pragma unroll 1
  for (int ct = 0; ct < NCT; ++ct) {
    const int buf = ct & 1;
    const bool has = (ct + 1 < NCT);

#if GE2E_USE_TDM
    if (wave == 0 && has)                    // async DMA next tile -> other buf
      tdm_load_tile(bpack + (size_t)(ct + 1) * 8192,
                    (unsigned)(size_t)(void*)&ldsb[buf ^ 1][0]);
#else
    uint4 r0, r1, r2, r3;
    if (has) {
      const uint4* s = (const uint4*)bpack + (size_t)(ct + 1) * 1024 + tid * 4;
      r0 = s[0]; r1 = s[1]; r2 = s[2]; r3 = s[3];
    }
#endif

    // ---- 16x16 tile of scores: 3-term bf16 WMMA over K=256, fragments
    //      software-pipelined one kt ahead so ds_load latency hides under WMMA
    v8f c = {};
    const v16bf* fb = (const v16bf*)&ldsb[buf][0];
    v16bf bh = fb[lane];
    v16bf bl = fb[32 + lane];
#pragma unroll
    for (int kt = 0; kt < NKT; ++kt) {
      v16bf bhn = bh, bln = bl;
      if (kt + 1 < NKT) {
        bhn = fb[((kt + 1) * 2 + 0) * 32 + lane];
        bln = fb[((kt + 1) * 2 + 1) * 32 + lane];
      }
      c = __builtin_amdgcn_wmma_f32_16x16x32_bf16(false, ah[kt], false, bh,
                                                  (short)0, c, false, false);
      c = __builtin_amdgcn_wmma_f32_16x16x32_bf16(false, al[kt], false, bh,
                                                  (short)0, c, false, false);
      c = __builtin_amdgcn_wmma_f32_16x16x32_bf16(false, ah[kt], false, bl,
                                                  (short)0, c, false, false);
      bh = bhn; bl = bln;
    }

    // ---- per-slot online logsumexp update (no cross-lane work in hot loop) --
    const bool isLab = (ct == labCT) && (ln == labLn);
#pragma unroll
    for (int v = 0; v < 8; ++v) {
      float x = fmaf(w, c[v], b);
      if (isLab) x = fmaf(w, dg[v], b);       // label-column substitution
      const float mn = fmaxf(mrun[v], x);
      srun[v] = srun[v] * __expf(mrun[v] - mn) + __expf(x - mn);
      mrun[v] = mn;
    }

#if GE2E_USE_TDM
    if (wave == 0 && has) __builtin_amdgcn_s_wait_tensorcnt(0);
#else
    if (has) {
      uint4* d = &ldsb[buf ^ 1][tid * 4];
      d[0] = r0; d[1] = r1; d[2] = r2; d[3] = r3;
    }
#endif
    __syncthreads();
  }

  // ---- merge 16 lane-partials per row, then sum (lse - target) over rows ----
  float local = 0.f;
#pragma unroll
  for (int v = 0; v < 8; ++v) {
    float m = mrun[v], s = srun[v];
#pragma unroll
    for (int off = 1; off <= 8; off <<= 1) {
      const float mo = __shfl_xor(m, off);
      const float so = __shfl_xor(s, off);
      const float mn = fmaxf(m, mo);
      s = s * __expf(m - mn) + so * __expf(mo - mn);
      m = mn;
    }
    local += m + __logf(s) - fmaf(w, dg[v], b);
  }
  local += __shfl_xor(local, 16);             // combine the two row halves
  if (lane == 0) part[rowTile] = local;       // deterministic: no atomics
}

// ------------------------------------------------------------------ finalize
__global__ __launch_bounds__(256) void k_fin(const float* __restrict__ part,
                                             float* __restrict__ out) {
  __shared__ double sm[256];
  double s = 0.0;
  for (int i = threadIdx.x; i < N_SPK; i += 256) s += (double)part[i];
  sm[threadIdx.x] = s;
  __syncthreads();
  for (int st = 128; st; st >>= 1) {
    if (threadIdx.x < st) sm[threadIdx.x] += sm[threadIdx.x + st];
    __syncthreads();
  }
  if (threadIdx.x == 0) out[0] = (float)(sm[0] * (1.0 / (double)NM));
}

extern "C" void kernel_launch(void* const* d_in, const int* in_sizes, int n_in,
                              void* d_out, int out_size, void* d_ws, size_t ws_size,
                              hipStream_t stream) {
  const float* emb = (const float*)d_in[0];
  const float* w   = (const float*)d_in[1];
  const float* b   = (const float*)d_in[2];
  float* out = (float*)d_out;

  char* ws = (char*)d_ws;
  float*  sums  = (float*)(ws + WS_SUMS);
  float*  diag  = (float*)(ws + WS_DIAG);
  float*  part  = (float*)(ws + WS_PART);
  __bf16* bpack = (__bf16*)(ws + WS_BPK);

  k_sums<<<N_SPK, 256, 0, stream>>>(emb, sums);
  k_pack<<<4096, 256, 0, stream>>>(sums, bpack);
  k_diag<<<4096, 256, 0, stream>>>(emb, sums, diag);
  k_main<<<256, 256, 0, stream>>>(emb, bpack, diag, w, b, part);
  k_fin<<<1, 256, 0, stream>>>(part, out);
}
